// HashHead_52733608460361
// MI455X (gfx1250) — matrix operand order repaired and checked by reading, not compiled, not run on hardware
//
#include <hip/hip_runtime.h>

// ---------------- problem constants ----------------
#define DDIM   1024
#define BITS   64
#define ROWS   32          // rows of X per block in pass1
#define XN_STRIDE 1032     // halves; pad 1024 -> 1032 (516-word row stride -> conflict-free b128 frag reads)
#define LSTR   68          // logits tile row stride (floats); 272B = 16B-aligned rows, stride%64 = 4
#define EPSV   1e-5f

typedef __attribute__((ext_vector_type(16))) _Float16 v16h;
typedef __attribute__((ext_vector_type(8)))  _Float16 v8h;
typedef __attribute__((ext_vector_type(4)))  _Float16 v4h;
typedef __attribute__((ext_vector_type(8)))  float    v8f;

union AFrag { v16h v; v8h h[2]; };

// ---------------- kernel 0: W fp32 -> f16 ----------------
__global__ void __launch_bounds__(256) convW_kernel(const float* __restrict__ W,
                                                    _Float16* __restrict__ Wh, int n) {
    int i = blockIdx.x * 256 + threadIdx.x;
    if (i < n) Wh[i] = (_Float16)W[i];
}

// ---------------- kernel 1: LN1 -> WMMA GEMM -> +bias -> LN2 -> store ----------------
__global__ void __launch_bounds__(256) pass1_kernel(
    const float* __restrict__ X,        // [N, D]
    const _Float16* __restrict__ Wh,    // [BITS, D] f16
    const float* __restrict__ bias,     // [BITS]
    const float* __restrict__ g1, const float* __restrict__ b1,   // [D]
    const float* __restrict__ g2, const float* __restrict__ b2,   // [BITS]
    float* __restrict__ logits_out,     // [N, BITS] normalized (post-LN2)
    float* __restrict__ blocksum)       // [numBlocks, BITS]
{
    extern __shared__ char smem[];
    _Float16* xn = (_Float16*)smem;                                      // ROWS x XN_STRIDE halves
    float*    lg = (float*)(smem + ROWS * XN_STRIDE * sizeof(_Float16)); // ROWS x LSTR floats

    const int tid  = threadIdx.x;
    const int wave = tid >> 5;
    const int lane = tid & 31;
    const long long rowBase = (long long)blockIdx.x * ROWS;

    // ---------- Phase 1: LayerNorm over D, write f16 tile to LDS ----------
    // 8 waves x 4 rows each; each lane covers 8 float4 chunks of the row.
    for (int rr = 0; rr < ROWS / 8; ++rr) {
        const int r = wave * (ROWS / 8) + rr;
        const float4* xrow = (const float4*)(X + (rowBase + r) * DDIM);
        float4 v[8];
        float s = 0.0f, ss = 0.0f;
        #pragma unroll
        for (int i = 0; i < 8; ++i) {
            v[i] = xrow[lane + i * 32];
            s  += v[i].x + v[i].y + v[i].z + v[i].w;
            ss += v[i].x * v[i].x + v[i].y * v[i].y + v[i].z * v[i].z + v[i].w * v[i].w;
        }
        #pragma unroll
        for (int off = 16; off > 0; off >>= 1) {
            s  += __shfl_xor(s,  off, 32);
            ss += __shfl_xor(ss, off, 32);
        }
        const float mu  = s * (1.0f / DDIM);
        const float var = ss * (1.0f / DDIM) - mu * mu;
        const float rs  = rsqrtf(var + EPSV);
        #pragma unroll
        for (int i = 0; i < 8; ++i) {
            const int c4 = lane + i * 32;            // float4 index along D
            const float4 gv = ((const float4*)g1)[c4];
            const float4 bv = ((const float4*)b1)[c4];
            v4h pk;
            pk[0] = (_Float16)((v[i].x - mu) * rs * gv.x + bv.x);
            pk[1] = (_Float16)((v[i].y - mu) * rs * gv.y + bv.y);
            pk[2] = (_Float16)((v[i].z - mu) * rs * gv.z + bv.z);
            pk[3] = (_Float16)((v[i].w - mu) * rs * gv.w + bv.w);
            *(v4h*)(xn + r * XN_STRIDE + c4 * 4) = pk;
        }
    }
    __syncthreads();

    // ---------- Phase 2: WMMA GEMM  logits[32 x 64] = xn[32 x 1024] * Wh^T ----------
    // 8 waves: 2 M-tiles x 4 N-tiles of 16x16.
    const int mt   = wave >> 2;            // 0..1
    const int nt   = wave & 3;             // 0..3
    const int hsel = lane >> 4;            // half-wave select
    const int l16  = lane & 15;

    const _Float16* arow = xn + (mt * 16 + l16) * XN_STRIDE;   // A: row = M
    const _Float16* brow = Wh + (nt * 16 + l16) * DDIM;        // B: column = bit -> W row

    v8f c = {};
    #pragma unroll 4
    for (int k = 0; k < DDIM; k += 32) {
        // A fragment (16-bit 16x32): lanes<16 -> K {0..7, 16..23}; lanes>=16 -> K {8..15, 24..31}
        AFrag a, b;
        const _Float16* ap = arow + k + hsel * 8;
        a.h[0] = *(const v8h*)(ap);
        a.h[1] = *(const v8h*)(ap + 16);
        // B fragment (16-bit 32x16): lanes<16 -> K {0..7, 16..23}; lanes>=16 -> K {8..15, 24..31}
        const _Float16* bp = brow + k + hsel * 8;
        b.h[0] = *(const v8h*)(bp);
        b.h[1] = *(const v8h*)(bp + 16);
        c = __builtin_amdgcn_wmma_f32_16x16x32_f16(false, a.v, false, b.v,
                                                   (short)0, c, false, false);
    }

    // C layout: VGPR v, lane l -> row = mt*16 + v + 8*(l>>4), col = nt*16 + (l&15)
    const int col   = nt * 16 + l16;
    const float bb  = bias[col];
    const int rbase = mt * 16 + 8 * hsel;
    #pragma unroll
    for (int v = 0; v < 8; ++v)
        lg[(rbase + v) * LSTR + col] = c[v] + bb;
    __syncthreads();

    // ---------- Phase 3: LN2 over BITS per row (one thread per row; LDS only) ----------
    if (tid < ROWS) {
        float* row = lg + tid * LSTR;
        float s = 0.0f, ss = 0.0f;
        #pragma unroll
        for (int j = 0; j < BITS; ++j) { float x = row[j]; s += x; ss += x * x; }
        const float mu  = s * (1.0f / BITS);
        const float var = ss * (1.0f / BITS) - mu * mu;
        const float rs  = rsqrtf(var + EPSV);
        #pragma unroll
        for (int j = 0; j < BITS; ++j)
            row[j] = (row[j] - mu) * rs * g2[j] + b2[j];
    }
    __syncthreads();

    // ---------- Phase 4a: cooperative coalesced float4 store of the 32x64 tile ----------
    {
        float4* out4 = (float4*)(logits_out + rowBase * BITS);   // 512 float4s, contiguous
        #pragma unroll
        for (int e = tid; e < ROWS * (BITS / 4); e += 256) {
            const int r  = e >> 4;          // row 0..31
            const int c4 = e & 15;          // float4 column 0..15
            out4[e] = *(const float4*)(lg + r * LSTR + c4 * 4);
        }
    }

    // ---------- Phase 4b: per-block column sums (deterministic, no FP atomics) ----------
    if (tid < BITS) {
        float s = 0.0f;
        #pragma unroll
        for (int r = 0; r < ROWS; ++r) s += lg[r * LSTR + tid];
        blocksum[(size_t)blockIdx.x * BITS + tid] = s;
    }
}

// ---------------- kernel 2: fixed-order reduction of block column sums ----------------
__global__ void __launch_bounds__(256) reduce_mean_kernel(const float* __restrict__ bsum,
                                                          float* __restrict__ mean,
                                                          int nb, float invN) {
    __shared__ float part[256];
    const int tid   = threadIdx.x;
    const int c     = tid & (BITS - 1);     // column 0..63
    const int chunk = tid >> 6;             // 0..3
    const int per   = nb >> 2;              // nb assumed multiple of 4
    float s = 0.0f;
    for (int i = chunk * per; i < (chunk + 1) * per; ++i)
        s += bsum[(size_t)i * BITS + c];
    part[tid] = s;
    __syncthreads();
    if (tid < BITS) {
        // fixed combine order -> bitwise deterministic
        float t = part[tid] + part[tid + 64] + part[tid + 128] + part[tid + 192];
        mean[tid] = t * invN;
    }
}

// ---------------- kernel 3: center -> tanh -> sign (float4 vectorized) ----------------
__global__ void __launch_bounds__(256) finalize_kernel(const float4* __restrict__ logits4,
                                                       const float* __restrict__ mean,
                                                       float4* __restrict__ H4,
                                                       float4* __restrict__ B4,
                                                       long long total4) {
    const long long i4 = (long long)blockIdx.x * 256 + threadIdx.x;
    if (i4 < total4) {
        const int c0 = (int)((i4 * 4) & (BITS - 1));   // multiple of 4, never wraps in a float4
        const float4 m = *(const float4*)(mean + c0);
        const float4 x = logits4[i4];
        float4 h, b;
        h.x = tanhf(0.5f * (x.x - m.x));
        h.y = tanhf(0.5f * (x.y - m.y));
        h.z = tanhf(0.5f * (x.z - m.z));
        h.w = tanhf(0.5f * (x.w - m.w));
        b.x = (h.x >= 0.0f) ? 1.0f : -1.0f;
        b.y = (h.y >= 0.0f) ? 1.0f : -1.0f;
        b.z = (h.z >= 0.0f) ? 1.0f : -1.0f;
        b.w = (h.w >= 0.0f) ? 1.0f : -1.0f;
        H4[i4] = h;
        B4[i4] = b;
    }
}

// ---------------- host launcher ----------------
extern "C" void kernel_launch(void* const* d_in, const int* in_sizes, int n_in,
                              void* d_out, int out_size, void* d_ws, size_t ws_size,
                              hipStream_t stream) {
    const float* X  = (const float*)d_in[0];
    const float* W  = (const float*)d_in[1];
    const float* b  = (const float*)d_in[2];
    const float* g1 = (const float*)d_in[3];
    const float* b1 = (const float*)d_in[4];
    const float* g2 = (const float*)d_in[5];
    const float* b2 = (const float*)d_in[6];

    const int N  = in_sizes[0] / DDIM;       // 65536
    const int NB = N / ROWS;                 // 2048 blocks
    const long long total = (long long)N * BITS;

    // workspace layout (256B aligned)
    char* ws = (char*)d_ws;
    _Float16* Wh     = (_Float16*)ws;                               // BITS*DDIM f16 = 128 KB
    size_t off = ((size_t)BITS * DDIM * sizeof(_Float16) + 255) & ~(size_t)255;
    float* logitsWS  = (float*)(ws + off);                          // N*BITS f32 = 16 MB
    off += ((size_t)N * BITS * sizeof(float) + 255) & ~(size_t)255;
    float* blocksum  = (float*)(ws + off);                          // NB*BITS f32
    off += ((size_t)NB * BITS * sizeof(float) + 255) & ~(size_t)255;
    float* meanWS    = (float*)(ws + off);                          // BITS f32

    float* H    = (float*)d_out;
    float* Bout = (float*)d_out + total;

    // 0) W -> f16
    convW_kernel<<<(BITS * DDIM + 255) / 256, 256, 0, stream>>>(W, Wh, BITS * DDIM);

    // 1) LN1 -> WMMA GEMM -> LN2 (dynamic LDS: 32x1032 f16 + 32x68 f32 = 74,752 B)
    const int smemBytes = ROWS * XN_STRIDE * sizeof(_Float16) + ROWS * LSTR * sizeof(float);
    pass1_kernel<<<NB, 256, smemBytes, stream>>>(X, Wh, b, g1, b1, g2, b2,
                                                 logitsWS, blocksum);

    // 2) deterministic column-mean reduction
    reduce_mean_kernel<<<1, 256, 0, stream>>>(blocksum, meanWS, NB, 1.0f / (float)N);

    // 3) center + tanh + binarize (float4)
    const long long total4 = total / 4;
    finalize_kernel<<<(int)((total4 + 255) / 256), 256, 0, stream>>>(
        (const float4*)logitsWS, meanWS, (float4*)H, (float4*)Bout, total4);
}
